// BertLayer_67834713473532
// MI455X (gfx1250) — compile-verified, matrix operand order checked by reading, not run on hardware
//
#include <hip/hip_runtime.h>
#include <math.h>

// ---------------------------------------------------------------------------
// BERT layer for MI455X (gfx1250): bf16 WMMA (f32 accumulate) everywhere,
// async global->LDS DMA (ASYNCcnt) double-buffering in the GEMMs.
// B=8 S=512 H=1024 NH=16 HD=64 FF=4096, M = B*S = 4096 tokens.
// ---------------------------------------------------------------------------

typedef __bf16 bf16_t;
typedef __attribute__((ext_vector_type(16))) __bf16 v16bf;
typedef __attribute__((ext_vector_type(8)))  float  v8f;

constexpr int Bc  = 8;
constexpr int Sc  = 512;
constexpr int Hc  = 1024;
constexpr int NHc = 16;
constexpr int HDc = 64;
constexpr int FFc = 4096;
constexpr int Mc  = Bc * Sc;   // 4096 token rows

__device__ __forceinline__ v8f wmma_bf16(v16bf a, v16bf b, v8f c) {
  // D = A(16x32 bf16) * B(32x16 bf16) + C(16x16 f32)
  return __builtin_amdgcn_wmma_f32_16x16x32_bf16(
      /*neg_a=*/false, a, /*neg_b=*/false, b,
      /*c_mod=*/(short)0, c, /*reuse_a=*/false, /*reuse_b=*/false);
}

// Wave-relative LDS byte offset of a __shared__ object: the flat address of a
// shared variable carries the LDS offset in its low 32 bits (LDS aperture
// truncates addr[31:0]).
__device__ __forceinline__ unsigned lds_off(const void* p) {
  return (unsigned)(uintptr_t)p;
}

// ---------------------------------------------------------------------------
// f32 -> bf16 conversion (RNE via clang __bf16 cast)
// ---------------------------------------------------------------------------
__global__ __launch_bounds__(256) void cvt_bf16(const float* __restrict__ in,
                                                bf16_t* __restrict__ out,
                                                int n) {
  int i = (blockIdx.x * 256 + threadIdx.x) * 4;
  if (i + 3 < n) {
#pragma unroll
    for (int j = 0; j < 4; ++j) out[i + j] = (bf16_t)in[i + j];
  }
}

// ---------------------------------------------------------------------------
// Tiled GEMM:  C[M,N] = A[M,K] @ W[N,K]^T + bias  (A,W bf16 K-contiguous)
// Block tile 128x128, K-step 32. 256 threads = 8 waves in a 2x4 wave grid,
// each wave owns a 64x32 tile = 4x2 WMMA accumulators.
//
// Global->LDS staging uses GLOBAL_LOAD_ASYNC_TO_LDS_B128 (ASYNCcnt) with
// double-buffered LDS tiles: stage k+1's DMA overlaps stage k's WMMAs, and
// no VGPRs touch the tile data.
//
// MODE 0: C -> bf16             (QKV projections)
// MODE 1: C -> f32              (attn-out dense, FF-out dense; LN fuses resid)
// MODE 2: C -> gelu_erf -> bf16 (FF intermediate)
// ---------------------------------------------------------------------------
template <int MODE>
__global__ __launch_bounds__(256)
void gemm_bf16(const bf16_t* __restrict__ A, const bf16_t* __restrict__ W,
               const float* __restrict__ bias, float* __restrict__ Cf,
               bf16_t* __restrict__ Cb, int Ndim, int K) {
  // +8 bf16 pad -> 80B row stride: keeps 16-elem fragment loads 16B aligned
  // and rows on distinct bank phases. Two buffers for the async pipeline.
  __shared__ bf16_t As[2][128][40];
  __shared__ bf16_t Bs[2][128][40];

  const int t    = threadIdx.x;
  const int lane = t & 31;
  const int w    = t >> 5;            // wave 0..7
  const int wm   = (w >> 2) * 64;     // wave M origin in block: 0 / 64
  const int wn   = (w & 3) * 32;      // wave N origin in block: 0..96
  const int m0   = blockIdx.y * 128;
  const int n0   = blockIdx.x * 128;

  const int lr = t >> 1;              // 0..127 : tile row this thread stages
  const int lc = (t & 1) * 16;        // 0 / 16 : 32B chunk within K-slab

  const int lrow = lane & 15;         // fragment row/col within 16
  const int lkh  = (lane >> 4) * 16;  // K half select (A/B fragment layout)

  // Per-thread base byte offsets for GVS addressing (saddr = matrix base).
  const unsigned a_goff = (unsigned)(((size_t)(m0 + lr) * K + lc) * 2);
  const unsigned b_goff = (unsigned)(((size_t)(n0 + lr) * K + lc) * 2);

  // Issue one K-slab's DMA: 4 x b128 per thread (2 per matrix; offset:16
  // advances both the global and the LDS address per the ISA pseudocode).
  auto issue = [&](int k0, int buf) {
    const unsigned al = lds_off(&As[buf][lr][lc]);
    const unsigned bl = lds_off(&Bs[buf][lr][lc]);
    const unsigned ag = a_goff + (unsigned)(k0 * 2);
    const unsigned bg = b_goff + (unsigned)(k0 * 2);
    asm volatile(
        "global_load_async_to_lds_b128 %0, %1, %2\n\t"
        "global_load_async_to_lds_b128 %0, %1, %2 offset:16\n\t"
        "global_load_async_to_lds_b128 %3, %4, %5\n\t"
        "global_load_async_to_lds_b128 %3, %4, %5 offset:16"
        :
        : "v"(al), "v"(ag), "s"(A), "v"(bl), "v"(bg), "s"(W)
        : "memory");
  };

  v8f acc[4][2];
#pragma unroll
  for (int i = 0; i < 4; ++i)
#pragma unroll
    for (int j = 0; j < 2; ++j) acc[i][j] = (v8f)0.0f;

  issue(0, 0);  // prologue: stage 0 in flight

  int s = 0;
  for (int k0 = 0; k0 < K; k0 += 32, s ^= 1) {
    // Current stage's DMA complete (ASYNCcnt in-order), then rendezvous so
    // every wave sees the LDS tile and is done reading the other buffer.
    asm volatile("s_wait_asynccnt 0x0" ::: "memory");
    __syncthreads();
    if (k0 + 32 < K) issue(k0 + 32, s ^ 1);
    if (k0 + 64 < K) {  // warm L2 two slabs ahead
      __builtin_prefetch(A + (size_t)(m0 + lr) * K + k0 + 64 + lc, 0, 0);
      __builtin_prefetch(W + (size_t)(n0 + lr) * K + k0 + 64 + lc, 0, 0);
    }

    v16bf af[4], bf2[2];
#pragma unroll
    for (int i = 0; i < 4; ++i)
      af[i] = *(const v16bf*)&As[s][wm + i * 16 + lrow][lkh];
#pragma unroll
    for (int j = 0; j < 2; ++j)
      bf2[j] = *(const v16bf*)&Bs[s][wn + j * 16 + lrow][lkh];
#pragma unroll
    for (int i = 0; i < 4; ++i)
#pragma unroll
      for (int j = 0; j < 2; ++j)
        acc[i][j] = wmma_bf16(af[i], bf2[j], acc[i][j]);
  }

  // Epilogue: D layout -> lane holds col (lane&15); VGPR r -> row r+8*(lane>>4)
  const int rsel = 8 * (lane >> 4);
#pragma unroll
  for (int j = 0; j < 2; ++j) {
    const int colg = n0 + wn + j * 16 + lrow;
    const float bv = bias[colg];
#pragma unroll
    for (int i = 0; i < 4; ++i) {
#pragma unroll
      for (int r = 0; r < 8; ++r) {
        const int rowg = m0 + wm + i * 16 + r + rsel;
        float val = acc[i][j][r] + bv;
        if (MODE == 1) {
          Cf[(size_t)rowg * Ndim + colg] = val;
        } else if (MODE == 2) {
          float g = 0.5f * val * (1.0f + erff(val / 1.41421f));
          Cb[(size_t)rowg * Ndim + colg] = (bf16_t)g;
        } else {
          Cb[(size_t)rowg * Ndim + colg] = (bf16_t)val;
        }
      }
    }
  }
}

// ---------------------------------------------------------------------------
// Attention: one wave per (b, h, 16 query rows). S=512, HD=64.
// scores = (Q K^T)/8 + mask  -> softmax (LDS) -> ctx = P V, all via WMMA.
// Q/K fragments load straight from global (head-dim contiguous). V is staged
// transposed in LDS 32 tokens at a time for the B-fragment of P@V.
// ---------------------------------------------------------------------------
__global__ __launch_bounds__(32)
void attention(const bf16_t* __restrict__ q, const bf16_t* __restrict__ k,
               const bf16_t* __restrict__ v, const float* __restrict__ mask,
               bf16_t* __restrict__ ctx) {
  __shared__ float  sc[16][516];   // f32 scores, pad 4 -> rows on distinct banks
  __shared__ bf16_t pb[16][528];   // bf16 exp(scores-max), A-frag-ready layout
  __shared__ bf16_t vt[64][40];    // V^T staging for one 32-token block
  __shared__ float  rinv[16];      // 1/rowsum

  const int lane = threadIdx.x;
  const int bh   = blockIdx.y;             // b*NH + h
  const int b    = bh / NHc;
  const int h    = bh - b * NHc;
  const int hoff = h * HDc;
  const int qt0  = blockIdx.x * 16;        // first query row of this wave
  const int lrow = lane & 15;
  const int half = lane >> 4;
  const int lkh  = half * 16;

  // --- Q fragments: A(16x32) x2 covering HD=64 -----------------------------
  const bf16_t* qb =
      q + (size_t)(b * Sc + qt0 + lrow) * Hc + hoff + lkh;
  v16bf qa0 = *(const v16bf*)(qb);
  v16bf qa1 = *(const v16bf*)(qb + 32);

  // --- scores: 32 column tiles of 16 keys, 2 WMMA each (K=64) --------------
  for (int n0 = 0; n0 < Sc; n0 += 16) {
    const bf16_t* kb =
        k + (size_t)(b * Sc + n0 + lrow) * Hc + hoff + lkh;
    v16bf kb0 = *(const v16bf*)(kb);
    v16bf kb1 = *(const v16bf*)(kb + 32);
    v8f s = (v8f)0.0f;
    s = wmma_bf16(qa0, kb0, s);
    s = wmma_bf16(qa1, kb1, s);
    const float mv = mask[b * Sc + n0 + lrow];   // [B,1,1,S] broadcast
#pragma unroll
    for (int r = 0; r < 8; ++r)
      sc[r + 8 * half][n0 + lrow] = s[r] * 0.125f + mv;  // 1/sqrt(64)
  }
  __syncthreads();

  // --- softmax: lane handles row (lane&15), column half (lane>>4) ----------
  {
    const int row = lrow, c0 = half * 256;
    float mx = -3.0e38f;
    for (int c = 0; c < 256; ++c) mx = fmaxf(mx, sc[row][c0 + c]);
    mx = fmaxf(mx, __shfl_xor(mx, 16));
    float sum = 0.0f;
    for (int c = 0; c < 256; ++c) {
      float e = __expf(sc[row][c0 + c] - mx);
      sum += e;
      pb[row][c0 + c] = (bf16_t)e;   // unnormalized; 1/sum folded in later
    }
    sum += __shfl_xor(sum, 16);
    rinv[row] = 1.0f / sum;          // both halves write identical value
  }
  __syncthreads();

  // --- ctx = P @ V : 16 K-blocks of 32 keys, 4 HD output tiles -------------
  v8f cacc[4];
#pragma unroll
  for (int jt = 0; jt < 4; ++jt) cacc[jt] = (v8f)0.0f;

  for (int kb = 0; kb < 16; ++kb) {
    // stage V^T for tokens [kb*32, kb*32+32): lane owns one token row
    const bf16_t* vr = v + (size_t)(b * Sc + kb * 32 + lane) * Hc + hoff;
    v16bf v0 = *(const v16bf*)(vr);
    v16bf v1 = *(const v16bf*)(vr + 16);
    v16bf v2 = *(const v16bf*)(vr + 32);
    v16bf v3 = *(const v16bf*)(vr + 48);
    __syncthreads();                 // previous block's WMMA reads done
#pragma unroll
    for (int d = 0; d < 16; ++d) {
      vt[d][lane]      = v0[d];
      vt[16 + d][lane] = v1[d];
      vt[32 + d][lane] = v2[d];
      vt[48 + d][lane] = v3[d];
    }
    __syncthreads();

    v16bf pa = *(const v16bf*)&pb[lrow][kb * 32 + lkh];  // A frag (16x32)
#pragma unroll
    for (int jt = 0; jt < 4; ++jt) {
      v16bf bfr = *(const v16bf*)&vt[jt * 16 + lrow][lkh];  // B frag (32x16)
      cacc[jt] = wmma_bf16(pa, bfr, cacc[jt]);
    }
  }

  // --- epilogue: scale by 1/rowsum, store ctx as [token, H] bf16 -----------
#pragma unroll
  for (int jt = 0; jt < 4; ++jt) {
#pragma unroll
    for (int r = 0; r < 8; ++r) {
      const int row = r + 8 * half;
      const float val = cacc[jt][r] * rinv[row];
      ctx[(size_t)(b * Sc + qt0 + row) * Hc + hoff + jt * 16 + lrow] =
          (bf16_t)val;
    }
  }
}

// ---------------------------------------------------------------------------
// Fused residual + LayerNorm over H=1024. One block per token row.
// Writes f32 (residual chain / final output) and optional bf16 (next GEMM A).
// ---------------------------------------------------------------------------
__global__ __launch_bounds__(256)
void resid_ln(const float* __restrict__ dense, const float* __restrict__ resid,
              const float* __restrict__ gamma, const float* __restrict__ beta,
              float* __restrict__ outf, bf16_t* __restrict__ outb) {
  const int row = blockIdx.x;
  const int t = threadIdx.x;
  const float* dr = dense + (size_t)row * Hc;
  const float* rr = resid + (size_t)row * Hc;

  float x[4];
  float s = 0.0f, qsum = 0.0f;
#pragma unroll
  for (int i = 0; i < 4; ++i) {
    x[i] = dr[t * 4 + i] + rr[t * 4 + i];
    s += x[i];
    qsum += x[i] * x[i];
  }
#pragma unroll
  for (int off = 16; off > 0; off >>= 1) {
    s += __shfl_xor(s, off);
    qsum += __shfl_xor(qsum, off);
  }
  __shared__ float ss[8], qq[8];
  if ((t & 31) == 0) { ss[t >> 5] = s; qq[t >> 5] = qsum; }
  __syncthreads();
  float S = 0.0f, Q = 0.0f;
#pragma unroll
  for (int i = 0; i < 8; ++i) { S += ss[i]; Q += qq[i]; }
  const float mu  = S * (1.0f / Hc);
  const float var = Q * (1.0f / Hc) - mu * mu;
  const float inv = rsqrtf(var + 1e-12f);
#pragma unroll
  for (int i = 0; i < 4; ++i) {
    const int c = t * 4 + i;
    const float y = (x[i] - mu) * inv * gamma[c] + beta[c];
    outf[(size_t)row * Hc + c] = y;
    if (outb) outb[(size_t)row * Hc + c] = (bf16_t)y;
  }
}

// ---------------------------------------------------------------------------
// Host-side orchestration
// ---------------------------------------------------------------------------
extern "C" void kernel_launch(void* const* d_in, const int* in_sizes, int n_in,
                              void* d_out, int out_size, void* d_ws,
                              size_t ws_size, hipStream_t stream) {
  (void)in_sizes; (void)n_in; (void)out_size; (void)ws_size;

  const float* x    = (const float*)d_in[0];
  const float* mask = (const float*)d_in[1];
  const float* Wq   = (const float*)d_in[2];
  const float* bq   = (const float*)d_in[3];
  const float* Wk   = (const float*)d_in[4];
  const float* bk   = (const float*)d_in[5];
  const float* Wv   = (const float*)d_in[6];
  const float* bv   = (const float*)d_in[7];
  const float* Wo   = (const float*)d_in[8];
  const float* bo   = (const float*)d_in[9];
  const float* g1   = (const float*)d_in[10];
  const float* b1   = (const float*)d_in[11];
  const float* Wi   = (const float*)d_in[12];
  const float* bi   = (const float*)d_in[13];
  const float* Wo2  = (const float*)d_in[14];
  const float* bo2  = (const float*)d_in[15];
  const float* g2   = (const float*)d_in[16];
  const float* b2   = (const float*)d_in[17];
  float* out = (float*)d_out;

  char* ws = (char*)d_ws;
  size_t off = 0;
  auto alloc = [&](size_t bytes) -> void* {
    void* p = ws + off;
    off += (bytes + 255) & ~(size_t)255;
    return p;
  };

  bf16_t* xb    = (bf16_t*)alloc((size_t)Mc * Hc * 2);
  bf16_t* Wqb   = (bf16_t*)alloc((size_t)Hc * Hc * 2);
  bf16_t* Wkb   = (bf16_t*)alloc((size_t)Hc * Hc * 2);
  bf16_t* Wvb   = (bf16_t*)alloc((size_t)Hc * Hc * 2);
  bf16_t* Wob   = (bf16_t*)alloc((size_t)Hc * Hc * 2);
  bf16_t* Wib   = (bf16_t*)alloc((size_t)FFc * Hc * 2);
  bf16_t* Wo2b  = (bf16_t*)alloc((size_t)Hc * FFc * 2);
  bf16_t* qb    = (bf16_t*)alloc((size_t)Mc * Hc * 2);
  bf16_t* kb    = (bf16_t*)alloc((size_t)Mc * Hc * 2);
  bf16_t* vb    = (bf16_t*)alloc((size_t)Mc * Hc * 2);
  bf16_t* ctxb  = (bf16_t*)alloc((size_t)Mc * Hc * 2);
  float*  attnD = (float*)alloc((size_t)Mc * Hc * 4);
  float*  y1f   = (float*)alloc((size_t)Mc * Hc * 4);
  bf16_t* y1b   = (bf16_t*)alloc((size_t)Mc * Hc * 2);
  bf16_t* inter = (bf16_t*)alloc((size_t)Mc * FFc * 2);
  float*  ffD   = (float*)alloc((size_t)Mc * Hc * 4);

  // ---- f32 -> bf16 conversions (per-launch; no cached state) --------------
  auto cvt = [&](const float* src, bf16_t* dst, int n) {
    cvt_bf16<<<n / 1024, 256, 0, stream>>>(src, dst, n);
  };
  cvt(x,   xb,   Mc * Hc);
  cvt(Wq,  Wqb,  Hc * Hc);
  cvt(Wk,  Wkb,  Hc * Hc);
  cvt(Wv,  Wvb,  Hc * Hc);
  cvt(Wo,  Wob,  Hc * Hc);
  cvt(Wi,  Wib,  FFc * Hc);
  cvt(Wo2, Wo2b, Hc * FFc);

  // ---- QKV projections: bf16 out -----------------------------------------
  {
    dim3 grid(Hc / 128, Mc / 128);
    gemm_bf16<0><<<grid, 256, 0, stream>>>(xb, Wqb, bq, nullptr, qb, Hc, Hc);
    gemm_bf16<0><<<grid, 256, 0, stream>>>(xb, Wkb, bk, nullptr, kb, Hc, Hc);
    gemm_bf16<0><<<grid, 256, 0, stream>>>(xb, Wvb, bv, nullptr, vb, Hc, Hc);
  }

  // ---- attention: grid (S/16 q-tiles, B*NH heads), one wave each ----------
  {
    dim3 grid(Sc / 16, Bc * NHc);
    attention<<<grid, 32, 0, stream>>>(qb, kb, vb, mask, ctxb);
  }

  // ---- attn output dense (f32) + residual/LN1 -----------------------------
  {
    dim3 grid(Hc / 128, Mc / 128);
    gemm_bf16<1><<<grid, 256, 0, stream>>>(ctxb, Wob, bo, attnD, nullptr, Hc,
                                           Hc);
    resid_ln<<<Mc, 256, 0, stream>>>(attnD, x, g1, b1, y1f, y1b);
  }

  // ---- FF intermediate: fused bias + erf-GELU, bf16 out -------------------
  {
    dim3 grid(FFc / 128, Mc / 128);
    gemm_bf16<2><<<grid, 256, 0, stream>>>(y1b, Wib, bi, nullptr, inter, FFc,
                                           Hc);
  }

  // ---- FF output dense (f32, K=4096) + residual/LN2 -> d_out --------------
  {
    dim3 grid(Hc / 128, Mc / 128);
    gemm_bf16<1><<<grid, 256, 0, stream>>>(inter, Wo2b, bo2, ffD, nullptr, Hc,
                                           FFc);
    resid_ln<<<Mc, 256, 0, stream>>>(ffD, y1f, g2, b2, out, nullptr);
  }
}
